// NHVSing_21680994910680
// MI455X (gfx1250) — matrix operand order repaired and checked by reading, compiled, not run on hardware
//
#include <hip/hip_runtime.h>
#include <math.h>

typedef __attribute__((ext_vector_type(16))) _Float16 v16h;
typedef __attribute__((ext_vector_type(8)))  float    v8f;

#define BB    2
#define TT    320
#define DD    80
#define CHN   256
#define NLAY  10
#define CCEPN 222
#define HOP   256
#define NFFT  1024
#define NSAMP (TT*HOP)     /* 81920 */
#define BT    (BB*TT)      /* 640   */
#define FSR   24000.0f
#define MAXH  200
#define TWO_PI 6.28318530717958647692f

// ---------------------------------------------------------------------------
// Pack x (B,T,D) -> f16 time-major [BT][96] (channel-contiguous, zero-padded)
// ---------------------------------------------------------------------------
__global__ void pack_x_kernel(const float* __restrict__ x, _Float16* __restrict__ xp) {
    int i = blockIdx.x * blockDim.x + threadIdx.x;      // over BT*96
    if (i >= BT * 96) return;
    int c  = i % 96;
    int bt = i / 96;
    xp[i] = (c < DD) ? (_Float16)x[bt * DD + c] : (_Float16)0.f;
}

// ---------------------------------------------------------------------------
// Pack weights (Cout,Cin,3) f32 -> WMMA A-fragment order f16:
//   Apack[((rt*nCb + cb)*3 + kt)*512 + lane*16 + j]
// A 16x32 f16 packing used here (contiguous K, same order as B): lane row = lane&15; K = (lane>>4)*16 + j
// ---------------------------------------------------------------------------
__global__ void pack_w_kernel(const float* __restrict__ W, _Float16* __restrict__ Apack,
                              int Cin, int Cout, int rowTiles, int nCb) {
    int i = blockIdx.x * blockDim.x + threadIdx.x;
    int total = rowTiles * nCb * 3 * 32 * 16;
    if (i >= total) return;
    int j    = i & 15;
    int lane = (i >> 4) & 31;
    int rest = i >> 9;
    int kt = rest % 3; rest /= 3;
    int cb = rest % nCb;
    int rt = rest / nCb;
    int r16 = lane & 15, hi = lane >> 4;
    int kk  = hi * 16 + j;
    int cin = cb * 32 + kk;
    int r   = rt * 16 + r16;
    float wv = (r < Cout && cin < Cin) ? W[((size_t)r * Cin + cin) * 3 + kt] : 0.f;
    Apack[i] = (_Float16)wv;
}

// ---------------------------------------------------------------------------
// Conv-as-GEMM, WMMA f16 16x16x32, f32 accumulate. Fully unrolled K reduction.
// Activations: time-major f16 [BT][CINP]. Weights: pre-packed A fragments.
// MODE 0: input layer (tanh)        -> OutF32/OutF16 [BT][COUT]
// MODE 1: residual (h + tanh(conv)) -> OutF32/OutF16 [BT][COUT]
// MODE 2: head (linear)             -> OutF32 = cep, [col][CCEPN], COUT padded
// One wave = one 16x16 output tile; 4 waves / block.
// ---------------------------------------------------------------------------
template <int CIN, int COUT, int MODE>
__global__ __launch_bounds__(128)
void conv_wmma_kernel(const _Float16* __restrict__ Act,
                      const float* __restrict__ ResF32,
                      const _Float16* __restrict__ Apack,
                      const float* __restrict__ Bias,
                      float* __restrict__ OutF32,
                      _Float16* __restrict__ OutF16,
                      int dil)
{
    constexpr int CINP = (CIN + 31) & ~31;
    constexpr int NCB  = CINP / 32;

    const int lane    = threadIdx.x & 31;
    const int wave    = threadIdx.x >> 5;
    const int colTile = blockIdx.x * 4 + wave;      // BT/16 = 40 tiles, gridDim.x=10
    const int rowTile = blockIdx.y;
    const int r16 = lane & 15;
    const int hi  = lane >> 4;

    const int col = colTile * 16 + r16;             // B/C/D column = lane&15
    const int bb  = col / TT;
    const int tt  = col - bb * TT;

    const _Float16* apBase = Apack + (size_t)rowTile * NCB * 3 * 512 + (size_t)lane * 16;

    v8f acc = {0.f, 0.f, 0.f, 0.f, 0.f, 0.f, 0.f, 0.f};

    #pragma unroll
    for (int cb = 0; cb < NCB; ++cb) {
        #pragma unroll
        for (int kt = 0; kt < 3; ++kt) {
            // A fragment: one aligned 32B vector load from packed weights
            v16h a = *(const v16h*)(apBase + ((size_t)(cb * 3 + kt)) * 512);
            // B fragment: 16 contiguous cin values at shifted time column
            const int ts = tt + dil * (kt - 1);
            v16h bf = {};
            if (ts >= 0 && ts < TT) {
                bf = *(const v16h*)(Act + (size_t)(bb * TT + ts) * CINP + cb * 32 + hi * 16);
            }
            acc = __builtin_amdgcn_wmma_f32_16x16x32_f16(
                      false, a, false, bf, (short)0, acc, false, false);
        }
    }

    // Epilogue: rows r = rowTile*16 + hi*8 + 0..7 are contiguous channels.
    const int r0 = rowTile * 16 + hi * 8;
    if (MODE == 2) {
        // heads: COUT padded to 224; mask the 222 tail, scalar writes
        #pragma unroll
        for (int v = 0; v < 8; ++v) {
            const int r = r0 + v;
            if (r < CCEPN)
                OutF32[(size_t)col * CCEPN + r] = acc[v] + Bias[r];
        }
    } else {
        float4 bias0 = *(const float4*)(Bias + r0);
        float4 bias1 = *(const float4*)(Bias + r0 + 4);
        float vals[8];
        vals[0] = tanhf(acc[0] + bias0.x); vals[1] = tanhf(acc[1] + bias0.y);
        vals[2] = tanhf(acc[2] + bias0.z); vals[3] = tanhf(acc[3] + bias0.w);
        vals[4] = tanhf(acc[4] + bias1.x); vals[5] = tanhf(acc[5] + bias1.y);
        vals[6] = tanhf(acc[6] + bias1.z); vals[7] = tanhf(acc[7] + bias1.w);
        if (MODE == 1) {
            float4 res0 = *(const float4*)(ResF32 + (size_t)col * COUT + r0);
            float4 res1 = *(const float4*)(ResF32 + (size_t)col * COUT + r0 + 4);
            vals[0] += res0.x; vals[1] += res0.y; vals[2] += res0.z; vals[3] += res0.w;
            vals[4] += res1.x; vals[5] += res1.y; vals[6] += res1.z; vals[7] += res1.w;
        }
        float* o32 = OutF32 + (size_t)col * COUT + r0;
        *(float4*)(o32)     = make_float4(vals[0], vals[1], vals[2], vals[3]);
        *(float4*)(o32 + 4) = make_float4(vals[4], vals[5], vals[6], vals[7]);
        _Float16 h8[8];
        #pragma unroll
        for (int v = 0; v < 8; ++v) h8[v] = (_Float16)vals[v];
        *(float4*)(OutF16 + (size_t)col * COUT + r0) = *(const float4*)h8;   // 16B store
    }
}

// ---------------------------------------------------------------------------
// F0 prefix sum per batch: P[b][t] = sum_{t'<t} cf0[b][t']
// ---------------------------------------------------------------------------
__global__ void f0_prefix_kernel(const float* __restrict__ cf0, float* __restrict__ P) {
    int b = blockIdx.x;
    if (threadIdx.x == 0) {
        float s = 0.f;
        for (int t = 0; t < TT; ++t) { P[b * TT + t] = s; s += cf0[b * TT + t]; }
    }
}

// ---------------------------------------------------------------------------
// Band-limited impulse train via Chebyshev cosine recurrence
// ---------------------------------------------------------------------------
__global__ void impulse_kernel(const float* __restrict__ cf0,
                               const float* __restrict__ P,
                               float* __restrict__ src)
{
    int i = blockIdx.x * blockDim.x + threadIdx.x;
    if (i >= BB * NSAMP) return;
    int b = i / NSAMP, n = i - b * NSAMP;
    int t = n >> 8, q = n & 255;
    float f0  = cf0[b * TT + t];
    float cum = P[b * TT + t] * (float)HOP + (float)(q + 1) * f0;
    float ph  = cum * (TWO_PI / FSR);
    ph -= floorf(ph * (1.f / TWO_PI)) * TWO_PI;

    int kmax = (int)ceilf((FSR * 0.5f) / f0) - 1;       // largest k with k*f0 < FS/2
    if (kmax > MAXH) kmax = MAXH;
    if (kmax < 0) kmax = 0;

    float c1 = __cosf(ph);
    float ckm1 = 1.f, ck = c1, s = 0.f;
    for (int k = 1; k <= kmax; ++k) {
        s += ck;
        float cn = 2.f * c1 * ck - ckm1;
        ckm1 = ck; ck = cn;
    }
    float voiced = (f0 > 0.f) ? 1.f : 0.f;
    src[i] = s * voiced * (1.f / (float)MAXH);
}

// ---------------------------------------------------------------------------
// Radix-2 Stockham FFT stages in LDS (n=1024, 10 stages, result ends in `a`)
// ---------------------------------------------------------------------------
__device__ inline void fft_stages(float2* a, float2* b, float sign, int tid) {
    float2* srcp = a;
    float2* dstp = b;
    int ncur = NFFT, s = 1;
    while (ncur > 1) {
        int m = ncur >> 1;
        float th0 = sign * TWO_PI / (float)ncur;
        for (int i = tid; i < NFFT / 2; i += 256) {
            int p = i / s;
            int q = i - p * s;
            float sn, cs;
            __sincosf(th0 * (float)p, &sn, &cs);
            float2 u = srcp[q + s * p];
            float2 w = srcp[q + s * (p + m)];
            float2 dif = {u.x - w.x, u.y - w.y};
            dstp[q + s * (2 * p)]     = {u.x + w.x, u.y + w.y};
            dstp[q + s * (2 * p + 1)] = {dif.x * cs - dif.y * sn, dif.x * sn + dif.y * cs};
        }
        __syncthreads();
        ncur = m; s <<= 1;
        float2* tsw = srcp; srcp = dstp; dstp = tsw;
    }
}

// ---------------------------------------------------------------------------
// ccep -> impulse response: imp = Re(ifft(exp(fft(zero-padded ccep))))
// One workgroup (256 thr) per frame; 1280 frames (harm then noise).
// ---------------------------------------------------------------------------
__global__ __launch_bounds__(256)
void ccep2imp_kernel(const float* __restrict__ cep, float* __restrict__ Imp) {
    __shared__ float2 sA[NFFT];
    __shared__ float2 sB[NFFT];
    const int f   = blockIdx.x;
    const int tid = threadIdx.x;
    const float* c = cep + (size_t)f * CCEPN;
    const int half = CCEPN / 2;                 // 111

    for (int q = tid; q < NFFT; q += 256) {
        float v = 0.f;
        if (q < half)              v = c[q];
        else if (q >= NFFT - half) v = c[half + q - (NFFT - half)];
        sA[q] = {v, 0.f};
    }
    __syncthreads();

    fft_stages(sA, sB, -1.f, tid);              // forward FFT -> sA

    for (int q = tid; q < NFFT; q += 256) {     // H = exp(C)
        float2 C = sA[q];
        float er = __expf(C.x);
        float sn, cs; __sincosf(C.y, &sn, &cs);
        sA[q] = {er * cs, er * sn};
    }
    __syncthreads();

    fft_stages(sA, sB, +1.f, tid);              // inverse FFT (unscaled) -> sA

    float* out = Imp + (size_t)f * NFFT;
    for (int q = tid; q < NFFT; q += 256)
        out[q] = sA[q].x * (1.f / (float)NFFT);
}

// ---------------------------------------------------------------------------
// LTV FIR + overlap-add (gather form) + sum both signals + clip
// ---------------------------------------------------------------------------
__global__ void ltv_oa_kernel(const float* __restrict__ harmSrc,
                              const float* __restrict__ z,
                              const float* __restrict__ Imp,
                              float* __restrict__ out)
{
    int i = blockIdx.x * blockDim.x + threadIdx.x;
    if (i >= BB * NSAMP) return;
    int b = i / NSAMP, n = i - b * NSAMP;

    int tHi = n >> 8;                if (tHi > TT - 1) tHi = TT - 1;
    int tLo = (n - 1279 + 255) >> 8; if (tLo < 0)      tLo = 0;

    float total = 0.f;
    for (int sig = 0; sig < 2; ++sig) {
        const float* srcB = (sig == 0 ? harmSrc : z) + (size_t)b * NSAMP;
        const float* impB = Imp + (size_t)(sig * BB + b) * TT * NFFT;
        float acc = 0.f;
        for (int t = tLo; t <= tHi; ++t) {
            int j = n - (t << 8);
            const float* fr = srcB + (t << 8);
            const float* im = impB + (size_t)t * NFFT;
            int iLo = j - (NFFT - 1); if (iLo < 0) iLo = 0;
            int iHi = (j < HOP - 1) ? j : (HOP - 1);
            for (int ii = iLo; ii <= iHi; ++ii)
                acc = fmaf(fr[ii], im[j - ii], acc);
        }
        total += acc;
    }
    total = fminf(1.f, fmaxf(-1.f, total));
    out[i] = total;
}

// ---------------------------------------------------------------------------
// Host-side orchestration
// ---------------------------------------------------------------------------
extern "C" void kernel_launch(void* const* d_in, const int* in_sizes, int n_in,
                              void* d_out, int out_size, void* d_ws, size_t ws_size,
                              hipStream_t stream) {
    const float* x     = (const float*)d_in[0];
    const float* cf0   = (const float*)d_in[1];
    const float* z     = (const float*)d_in[2];
    const float* w_in  = (const float*)d_in[3];
    const float* b_in  = (const float*)d_in[4];
    const float* w_lay = (const float*)d_in[5];
    const float* b_lay = (const float*)d_in[6];
    const float* w_h   = (const float*)d_in[7];
    const float* b_h   = (const float*)d_in[8];
    const float* w_n   = (const float*)d_in[9];
    const float* b_n   = (const float*)d_in[10];
    (void)in_sizes; (void)n_in; (void)out_size; (void)ws_size;

    // ---- carve workspace ----
    char* p = (char*)d_ws;
    auto carve = [&](size_t bytes) -> void* {
        void* r = (void*)p;
        p += (bytes + 255) & ~(size_t)255;
        return r;
    };
    _Float16* xp   = (_Float16*)carve((size_t)BT * 96 * sizeof(_Float16));
    float*    hAf  = (float*)   carve((size_t)BT * CHN * sizeof(float));
    _Float16* hAh  = (_Float16*)carve((size_t)BT * CHN * sizeof(_Float16));
    float*    hBf  = (float*)   carve((size_t)BT * CHN * sizeof(float));
    _Float16* hBh  = (_Float16*)carve((size_t)BT * CHN * sizeof(_Float16));
    float*    cep  = (float*)   carve((size_t)2 * BT * CCEPN * sizeof(float));
    float*    Ppre = (float*)   carve((size_t)BB * TT * sizeof(float));
    float*    hsrc = (float*)   carve((size_t)BB * NSAMP * sizeof(float));
    float*    Imp  = (float*)   carve((size_t)2 * BT * NFFT * sizeof(float));
    // packed weight fragments
    const size_t AP_IN  = (size_t)16 * 3 * 3 * 512;     // rowTiles*nCb*kt*512
    const size_t AP_LAY = (size_t)16 * 8 * 3 * 512;
    const size_t AP_HD  = (size_t)14 * 8 * 3 * 512;
    _Float16* apIn  = (_Float16*)carve(AP_IN * sizeof(_Float16));
    _Float16* apLay = (_Float16*)carve(AP_LAY * NLAY * sizeof(_Float16));
    _Float16* apH   = (_Float16*)carve(AP_HD * sizeof(_Float16));
    _Float16* apN   = (_Float16*)carve(AP_HD * sizeof(_Float16));

    // ---- 1) packing ----
    pack_x_kernel<<<(BT * 96 + 255) / 256, 256, 0, stream>>>(x, xp);
    pack_w_kernel<<<((int)AP_IN + 255) / 256, 256, 0, stream>>>(w_in, apIn, DD, CHN, 16, 3);
    for (int l = 0; l < NLAY; ++l)
        pack_w_kernel<<<((int)AP_LAY + 255) / 256, 256, 0, stream>>>(
            w_lay + (size_t)l * CHN * CHN * 3, apLay + AP_LAY * l, CHN, CHN, 16, 8);
    pack_w_kernel<<<((int)AP_HD + 255) / 256, 256, 0, stream>>>(w_h, apH, CHN, CCEPN, 14, 8);
    pack_w_kernel<<<((int)AP_HD + 255) / 256, 256, 0, stream>>>(w_n, apN, CHN, CCEPN, 14, 8);

    // ---- 2) conv stack (WMMA GEMMs) ----
    dim3 gFull(10, CHN / 16);
    conv_wmma_kernel<DD, CHN, 0><<<gFull, 128, 0, stream>>>(
        xp, nullptr, apIn, b_in, hAf, hAh, 1);
    const _Float16* curH = hAh;
    const float*    curF = hAf;
    for (int l = 0; l < NLAY; ++l) {
        float*    nxtF = (l & 1) ? hAf : hBf;
        _Float16* nxtH = (l & 1) ? hAh : hBh;
        conv_wmma_kernel<CHN, CHN, 1><<<gFull, 128, 0, stream>>>(
            curH, curF, apLay + AP_LAY * l, b_lay + (size_t)l * CHN,
            nxtF, nxtH, 1 << (l % 5));
        curH = nxtH; curF = nxtF;
    }
    dim3 gHead(10, 14);
    conv_wmma_kernel<CHN, 224, 2><<<gHead, 128, 0, stream>>>(
        curH, nullptr, apH, b_h, cep, nullptr, 1);
    conv_wmma_kernel<CHN, 224, 2><<<gHead, 128, 0, stream>>>(
        curH, nullptr, apN, b_n, cep + (size_t)BT * CCEPN, nullptr, 1);

    // ---- 3) DSP chain ----
    f0_prefix_kernel<<<BB, 32, 0, stream>>>(cf0, Ppre);
    impulse_kernel<<<(BB * NSAMP + 255) / 256, 256, 0, stream>>>(cf0, Ppre, hsrc);
    ccep2imp_kernel<<<2 * BT, 256, 0, stream>>>(cep, Imp);
    ltv_oa_kernel<<<(BB * NSAMP + 255) / 256, 256, 0, stream>>>(hsrc, z, Imp,
                                                                (float*)d_out);
}
